// EvolutionBlock_61976378081405
// MI455X (gfx1250) — matrix operand order repaired and checked by reading, not compiled, not run on hardware
//
#include <hip/hip_runtime.h>
#include <hip/hip_bf16.h>
#include <math.h>
#include <stdint.h>

// ---------------- types ----------------
typedef __bf16 bf16_t;
typedef __attribute__((ext_vector_type(16))) __bf16 v16bf;
typedef __attribute__((ext_vector_type(8)))  float  v8f;
typedef __attribute__((ext_vector_type(4)))  float  v4f;
typedef __attribute__((ext_vector_type(4)))  unsigned int u32x4;
typedef __attribute__((ext_vector_type(8)))  int    i32x8;
typedef __attribute__((ext_vector_type(4)))  int    i32x4;

// ---------------- problem constants ----------------
#define MTOK 8192     // B*T
#define DDIM 1024
#define HDFF 4096
#define SDIM 1024
#define KCONV 4
#define NEXP 8
#define HEXP 512
#define TSEQ 2048

#if defined(__HIP_DEVICE_COMPILE__) && __has_builtin(__builtin_amdgcn_tensor_load_to_lds) && __has_builtin(__builtin_amdgcn_s_wait_tensorcnt)
#define USE_TDM 1
#else
#define USE_TDM 0
#endif

// ---------------- helpers ----------------
__device__ __forceinline__ bf16_t f32_to_bf16(float f) {
  unsigned u = __builtin_bit_cast(unsigned, f);
  u += 0x7FFFu + ((u >> 16) & 1u);           // round-to-nearest-even
  unsigned short s = (unsigned short)(u >> 16);
  return __builtin_bit_cast(bf16_t, s);
}
__device__ __forceinline__ float silu(float x) { return x / (1.f + __expf(-x)); }

__device__ __forceinline__ v8f wmma_bf16(v16bf a, v16bf b, v8f c) {
  return __builtin_amdgcn_wmma_f32_16x16x32_bf16(
      false, a, false, b, (short)0, c, false, false);
}

// A fragment (16x32 bf16): lane = M row (lane&15); halves 0-7 at K=+8*(lane>>4),
// halves 8-15 at K=+16+8*(lane>>4)   (ISA 7.12.2, 16-bit A 16x32)
__device__ __forceinline__ v16bf load_a(const bf16_t* A, int lda, int mBase, int kBase) {
  int lane = threadIdx.x & 31;
  const bf16_t* r = A + (size_t)(mBase + (lane & 15)) * lda + kBase + (lane >> 4) * 8;
  union { v16bf v; v4f f[2]; } u;
  u.f[0] = *(const v4f*)r;
  u.f[1] = *(const v4f*)(r + 16);
  return u.v;
}
// A fragment with causal row shift + zero fill (conv taps)
__device__ __forceinline__ v16bf load_a_shift(const bf16_t* A, int lda, int mBase,
                                              int kBase, int shift, int T) {
  int lane = threadIdx.x & 31;
  int m = mBase + (lane & 15);
  bool valid = ((m % T) + shift) >= 0;
  union { v16bf v; v4f f[2]; } u;
  v4f z = {0.f, 0.f, 0.f, 0.f};
  if (valid) {
    const bf16_t* r = A + (size_t)(m + shift) * lda + kBase + (lane >> 4) * 8;
    u.f[0] = *(const v4f*)r;
    u.f[1] = *(const v4f*)(r + 16);
  } else {
    u.f[0] = z; u.f[1] = z;
  }
  return u.v;
}
// B fragment (32x16 bf16) from row-major W[N,K]: lane = N col (lane&15);
// halves 0-7 at K=+8*(lane>>4), halves 8-15 at K=+16+8*(lane>>4)
__device__ __forceinline__ v16bf load_b(const bf16_t* W, int ldw, int nBase, int kBase) {
  int lane = threadIdx.x & 31;
  const bf16_t* r = W + (size_t)(nBase + (lane & 15)) * ldw + kBase + (lane >> 4) * 8;
  union { v16bf v; v4f f[2]; } u;
  u.f[0] = *(const v4f*)r;
  u.f[1] = *(const v4f*)(r + 16);
  return u.v;
}

// 32x32 wave-tile core, register double-buffered (peeled last step).
// c[0]=(m0..15,n0..15) c[1]=(m0..15,n16..31) c[2]=(m16..31,n0..15) c[3]=(m16..31,n16..31)
__device__ __forceinline__ void gemm_core(const bf16_t* A, int lda, int mBase,
                                          const bf16_t* W, int ldw, int nBase,
                                          int Kdim, v8f (&c)[4]) {
  v16bf a0 = load_a(A, lda, mBase, 0);
  v16bf a1 = load_a(A, lda, mBase + 16, 0);
  v16bf b0 = load_b(W, ldw, nBase, 0);
  v16bf b1 = load_b(W, ldw, nBase + 16, 0);
  for (int k = 32; k < Kdim; k += 32) {
    v16bf a0n = load_a(A, lda, mBase, k);
    v16bf a1n = load_a(A, lda, mBase + 16, k);
    v16bf b0n = load_b(W, ldw, nBase, k);
    v16bf b1n = load_b(W, ldw, nBase + 16, k);
    c[0] = wmma_bf16(a0, b0, c[0]);
    c[1] = wmma_bf16(a0, b1, c[1]);
    c[2] = wmma_bf16(a1, b0, c[2]);
    c[3] = wmma_bf16(a1, b1, c[3]);
    a0 = a0n; a1 = a1n; b0 = b0n; b1 = b1n;
  }
  c[0] = wmma_bf16(a0, b0, c[0]);
  c[1] = wmma_bf16(a0, b1, c[1]);
  c[2] = wmma_bf16(a1, b0, c[2]);
  c[3] = wmma_bf16(a1, b1, c[3]);
}

// ---------------- conversion kernels ----------------
__global__ void k_cvt(const float* __restrict__ in, bf16_t* __restrict__ out, long long n) {
  long long i = (long long)blockIdx.x * blockDim.x + threadIdx.x;
  if (i < n) out[i] = f32_to_bf16(in[i]);
}
// sW_conv [S,S,K] f32 -> [K,S,S] bf16
__global__ void k_conv_repack(const float* __restrict__ in, bf16_t* __restrict__ out,
                              int S, int K) {
  long long n = (long long)S * S * K;
  long long i = (long long)blockIdx.x * blockDim.x + threadIdx.x;
  if (i >= n) return;
  int j = (int)(i % K);
  long long rem = i / K;
  int si = (int)(rem % S);
  int so = (int)(rem / S);
  out[(long long)j * S * S + (long long)so * S + si] = f32_to_bf16(in[i]);
}

// ---------------- routers ----------------
__global__ void k_router(const float* __restrict__ x, const float* __restrict__ rW,
                         const float* __restrict__ rb, float* __restrict__ bw,
                         int M, int D) {
  int row = blockIdx.x * (blockDim.x >> 5) + (threadIdx.x >> 5);
  if (row >= M) return;
  int lane = threadIdx.x & 31;
  const float* xr = x + (size_t)row * D;
  float s0 = 0.f, s1 = 0.f, s2 = 0.f;
  for (int i = lane; i < D; i += 32) {
    float xv = xr[i];
    s0 += xv * rW[i];
    s1 += xv * rW[D + i];
    s2 += xv * rW[2 * D + i];
  }
  for (int off = 16; off > 0; off >>= 1) {
    s0 += __shfl_xor(s0, off);
    s1 += __shfl_xor(s1, off);
    s2 += __shfl_xor(s2, off);
  }
  if (lane == 0) {
    s0 += rb[0]; s1 += rb[1]; s2 += rb[2];
    float mx = fmaxf(s0, fmaxf(s1, s2));
    float e0 = __expf(s0 - mx), e1 = __expf(s1 - mx), e2 = __expf(s2 - mx);
    float inv = 1.f / (e0 + e1 + e2);
    bw[(size_t)row * 3 + 0] = e0 * inv;
    bw[(size_t)row * 3 + 1] = e1 * inv;
    bw[(size_t)row * 3 + 2] = e2 * inv;
  }
}

__global__ void k_moe_router(const float* __restrict__ x, const float* __restrict__ mW,
                             const float* __restrict__ mb, const float* __restrict__ bw,
                             float* __restrict__ combine, int M, int D) {
  int row = blockIdx.x * (blockDim.x >> 5) + (threadIdx.x >> 5);
  if (row >= M) return;
  int lane = threadIdx.x & 31;
  const float* xr = x + (size_t)row * D;
  float s[NEXP];
#pragma unroll
  for (int e = 0; e < NEXP; ++e) s[e] = 0.f;
  for (int i = lane; i < D; i += 32) {
    float xv = xr[i];
#pragma unroll
    for (int e = 0; e < NEXP; ++e) s[e] += xv * mW[(size_t)e * D + i];
  }
  for (int off = 16; off > 0; off >>= 1) {
#pragma unroll
    for (int e = 0; e < NEXP; ++e) s[e] += __shfl_xor(s[e], off);
  }
  if (lane == 0) {
#pragma unroll
    for (int e = 0; e < NEXP; ++e) s[e] += mb[e];
    int i1 = 0; float v1 = s[0];
#pragma unroll
    for (int e = 1; e < NEXP; ++e) if (s[e] > v1) { v1 = s[e]; i1 = e; }
    int i2 = -1; float v2 = -3.4e38f;
#pragma unroll
    for (int e = 0; e < NEXP; ++e) if (e != i1 && s[e] > v2) { v2 = s[e]; i2 = e; }
    float mx = fmaxf(v1, v2);
    float e1 = __expf(v1 - mx), e2 = __expf(v2 - mx);
    float inv = 1.f / (e1 + e2);
    float g = bw[(size_t)row * 3 + 2];
    float* c = combine + (size_t)row * NEXP;
#pragma unroll
    for (int e = 0; e < NEXP; ++e) c[e] = 0.f;
    c[i1] = g * e1 * inv;
    c[i2] = g * e2 * inv;
  }
}

// ---------------- fused dual-GEMM swiglu (32M x 16N per half, A reused) ----------------
__global__ void k_gemm_swiglu(const bf16_t* __restrict__ A, int Kdim,
                              const bf16_t* __restrict__ W, const float* __restrict__ bias,
                              int Nhalf, bf16_t* __restrict__ Out) {
  int wave = threadIdx.x >> 5;
  int mBase = (blockIdx.x * 4 + wave) * 32;
  int nBase = blockIdx.y * 16;
  v8f ca0 = {}, ca1 = {}, cb0 = {}, cb1 = {};
  v16bf a0 = load_a(A, Kdim, mBase, 0);
  v16bf a1 = load_a(A, Kdim, mBase + 16, 0);
  v16bf wa = load_b(W, Kdim, nBase, 0);
  v16bf wb = load_b(W, Kdim, Nhalf + nBase, 0);
  for (int k = 32; k < Kdim; k += 32) {
    v16bf a0n = load_a(A, Kdim, mBase, k);
    v16bf a1n = load_a(A, Kdim, mBase + 16, k);
    v16bf wan = load_b(W, Kdim, nBase, k);
    v16bf wbn = load_b(W, Kdim, Nhalf + nBase, k);
    ca0 = wmma_bf16(a0, wa, ca0);
    ca1 = wmma_bf16(a1, wa, ca1);
    cb0 = wmma_bf16(a0, wb, cb0);
    cb1 = wmma_bf16(a1, wb, cb1);
    a0 = a0n; a1 = a1n; wa = wan; wb = wbn;
  }
  ca0 = wmma_bf16(a0, wa, ca0);
  ca1 = wmma_bf16(a1, wa, ca1);
  cb0 = wmma_bf16(a0, wb, cb0);
  cb1 = wmma_bf16(a1, wb, cb1);

  int lane = threadIdx.x & 31;
  int h = lane >> 4, col = lane & 15;
  float ba = bias[nBase + col];
  float bb = bias[Nhalf + nBase + col];
#pragma unroll
  for (int r = 0; r < 8; ++r) {
    int r0 = mBase + 8 * h + r;
    int r1 = r0 + 16;
    Out[(size_t)r0 * Nhalf + nBase + col] = f32_to_bf16(silu(ca0[r] + ba) * (cb0[r] + bb));
    Out[(size_t)r1 * Nhalf + nBase + col] = f32_to_bf16(silu(ca1[r] + ba) * (cb1[r] + bb));
  }
}

// ---------------- plain GEMM + bias -> bf16, B staged via TDM->LDS ----------------
__global__ void k_gemm_bias_tdm(const bf16_t* __restrict__ A, int Kdim,
                                const bf16_t* __restrict__ W, const float* __restrict__ bias,
                                int N, bf16_t* __restrict__ Out) {
  __shared__ __align__(16) bf16_t sB[2][32 * 32];   // double-buffered 32n x 32k tile
  int wave = threadIdx.x >> 5;
  int mBase = (blockIdx.x * 4 + wave) * 32;
  int nBase = blockIdx.y * 32;

#if USE_TDM
  unsigned ldsAddr[2] = { (unsigned)(uintptr_t)&sB[0][0], (unsigned)(uintptr_t)&sB[1][0] };
  auto stage = [&](int k, int buf) {
    if (threadIdx.x < 32) {   // one wave launches the DMA (EXEC ignored by TDM)
      unsigned long long ga = (unsigned long long)(uintptr_t)(W + (size_t)nBase * Kdim + k);
      u32x4 g0;
      g0[0] = 1u;                                              // count=1, user descriptor
      g0[1] = ldsAddr[buf];                                    // lds_addr
      g0[2] = (unsigned)(ga & 0xFFFFFFFFull);                  // global_addr[31:0]
      g0[3] = (unsigned)((ga >> 32) & 0x01FFFFFFull) | 0x80000000u; // addr[56:32] | type=2
      i32x8 g1;
      g1[0] = 0x00010000;                                      // data_size = 2 bytes
      g1[1] = (int)((unsigned)Kdim << 16);                     // tensor_dim0[15:0] @ [63:48]
      g1[2] = (int)(((unsigned)N & 0xFFFFu) << 16);            // tensor_dim1[15:0] @ [95:80]
      g1[3] = (int)(32u << 16);                                // tile_dim0=32 @ [127:112]
      g1[4] = 32;                                              // tile_dim1=32 @ [143:128]
      g1[5] = Kdim;                                            // tensor_dim0_stride[31:0]
      g1[6] = 0;
      g1[7] = 0;
      i32x4 gz4 = {0, 0, 0, 0};
      i32x8 gz8 = {0, 0, 0, 0, 0, 0, 0, 0};
      // 6-arg toolchain signature: (g0, g1, g2, g3, g_extra, cpol)
      __builtin_amdgcn_tensor_load_to_lds(g0, g1, gz4, gz4, gz8, 0);
    }
  };
#define STAGE_WAIT() do { if (threadIdx.x < 32) __builtin_amdgcn_s_wait_tensorcnt(0); } while (0)
#else
  auto stage = [&](int k, int buf) {
    int t = threadIdx.x;                 // 128 threads: 32 rows x 4 chunks of 8 bf16
    int row = t >> 2, ch = t & 3;
    *(v4f*)&sB[buf][row * 32 + ch * 8] =
        *(const v4f*)(W + (size_t)(nBase + row) * Kdim + k + ch * 8);
  };
#define STAGE_WAIT() do { } while (0)
#endif

  stage(0, 0);
  STAGE_WAIT();
  __syncthreads();

  v8f c[4] = {};
  v16bf a0 = load_a(A, Kdim, mBase, 0);
  v16bf a1 = load_a(A, Kdim, mBase + 16, 0);
  int buf = 0;
  for (int k = 0; k < Kdim; k += 32, buf ^= 1) {
    bool more = (k + 32) < Kdim;
    if (more) stage(k + 32, buf ^ 1);
    v16bf b0 = load_b(sB[buf], 32, 0, 0);   // ds_load fragments
    v16bf b1 = load_b(sB[buf], 32, 16, 0);
    v16bf a0n = a0, a1n = a1;
    if (more) {
      a0n = load_a(A, Kdim, mBase, k + 32);
      a1n = load_a(A, Kdim, mBase + 16, k + 32);
    }
    c[0] = wmma_bf16(a0, b0, c[0]);
    c[1] = wmma_bf16(a0, b1, c[1]);
    c[2] = wmma_bf16(a1, b0, c[2]);
    c[3] = wmma_bf16(a1, b1, c[3]);
    a0 = a0n; a1 = a1n;
    STAGE_WAIT();
    __syncthreads();
  }

  int lane = threadIdx.x & 31;
  int h = lane >> 4, col = lane & 15;
  float blo = bias[nBase + col], bhi = bias[nBase + 16 + col];
#pragma unroll
  for (int r = 0; r < 8; ++r) {
    int r0 = mBase + 8 * h + r;
    int r1 = r0 + 16;
    size_t o0 = (size_t)r0 * N + nBase + col;
    size_t o1 = (size_t)r1 * N + nBase + col;
    Out[o0]      = f32_to_bf16(c[0][r] + blo);
    Out[o0 + 16] = f32_to_bf16(c[1][r] + bhi);
    Out[o1]      = f32_to_bf16(c[2][r] + blo);
    Out[o1 + 16] = f32_to_bf16(c[3][r] + bhi);
  }
}

// ---------------- causal conv as 4 shifted GEMMs ----------------
__global__ void k_gemm_conv(const bf16_t* __restrict__ Hin, const bf16_t* __restrict__ Wc,
                            const float* __restrict__ bias, bf16_t* __restrict__ Out,
                            int S, int T) {
  int wave = threadIdx.x >> 5;
  int mBase = (blockIdx.x * 4 + wave) * 32;
  int nBase = blockIdx.y * 32;
  v8f c[4] = {};
#pragma unroll
  for (int j = 0; j < KCONV; ++j) {
    int shift = j - (KCONV - 1);
    const bf16_t* Wj = Wc + (size_t)j * S * S;
    v16bf a0 = load_a_shift(Hin, S, mBase, 0, shift, T);
    v16bf a1 = load_a_shift(Hin, S, mBase + 16, 0, shift, T);
    v16bf b0 = load_b(Wj, S, nBase, 0);
    v16bf b1 = load_b(Wj, S, nBase + 16, 0);
    for (int k = 32; k < S; k += 32) {
      v16bf a0n = load_a_shift(Hin, S, mBase, k, shift, T);
      v16bf a1n = load_a_shift(Hin, S, mBase + 16, k, shift, T);
      v16bf b0n = load_b(Wj, S, nBase, k);
      v16bf b1n = load_b(Wj, S, nBase + 16, k);
      c[0] = wmma_bf16(a0, b0, c[0]);
      c[1] = wmma_bf16(a0, b1, c[1]);
      c[2] = wmma_bf16(a1, b0, c[2]);
      c[3] = wmma_bf16(a1, b1, c[3]);
      a0 = a0n; a1 = a1n; b0 = b0n; b1 = b1n;
    }
    c[0] = wmma_bf16(a0, b0, c[0]);
    c[1] = wmma_bf16(a0, b1, c[1]);
    c[2] = wmma_bf16(a1, b0, c[2]);
    c[3] = wmma_bf16(a1, b1, c[3]);
  }
  int lane = threadIdx.x & 31;
  int h = lane >> 4, col = lane & 15;
  float blo = bias[nBase + col], bhi = bias[nBase + 16 + col];
#pragma unroll
  for (int r = 0; r < 8; ++r) {
    int r0 = mBase + 8 * h + r;
    int r1 = r0 + 16;
    size_t o0 = (size_t)r0 * S + nBase + col;
    size_t o1 = (size_t)r1 * S + nBase + col;
    Out[o0]      = f32_to_bf16(c[0][r] + blo);
    Out[o0 + 16] = f32_to_bf16(c[1][r] + bhi);
    Out[o1]      = f32_to_bf16(c[2][r] + blo);
    Out[o1 + 16] = f32_to_bf16(c[3][r] + bhi);
  }
}

// ---------------- GEMM + bias, per-row scale, accumulate into f32 out ----------------
__global__ void k_gemm_scaled_acc(const bf16_t* __restrict__ A, int Kdim,
                                  const bf16_t* __restrict__ W, const float* __restrict__ bias,
                                  int N, const float* __restrict__ scale, int sStride, int sOff,
                                  float* __restrict__ Out, int accumulate) {
  int wave = threadIdx.x >> 5;
  int mBase = (blockIdx.x * 4 + wave) * 32;
  int nBase = blockIdx.y * 32;
  v8f c[4] = {};
  gemm_core(A, Kdim, mBase, W, Kdim, nBase, Kdim, c);

  int lane = threadIdx.x & 31;
  int h = lane >> 4, col = lane & 15;
  float blo = bias[nBase + col], bhi = bias[nBase + 16 + col];
#pragma unroll
  for (int r = 0; r < 8; ++r) {
    int r0 = mBase + 8 * h + r;
    int r1 = r0 + 16;
    float s0 = scale[(size_t)r0 * sStride + sOff];
    float s1 = scale[(size_t)r1 * sStride + sOff];
    size_t o0 = (size_t)r0 * N + nBase + col;
    size_t o1 = (size_t)r1 * N + nBase + col;
    float p00 = accumulate ? Out[o0]      : 0.f;
    float p01 = accumulate ? Out[o0 + 16] : 0.f;
    float p10 = accumulate ? Out[o1]      : 0.f;
    float p11 = accumulate ? Out[o1 + 16] : 0.f;
    Out[o0]      = p00 + s0 * (c[0][r] + blo);
    Out[o0 + 16] = p01 + s0 * (c[1][r] + bhi);
    Out[o1]      = p10 + s1 * (c[2][r] + blo);
    Out[o1 + 16] = p11 + s1 * (c[3][r] + bhi);
  }
}

// ---------------- host-side launcher ----------------
extern "C" void kernel_launch(void* const* d_in, const int* in_sizes, int n_in,
                              void* d_out, int out_size, void* d_ws, size_t ws_size,
                              hipStream_t stream) {
  (void)in_sizes; (void)n_in; (void)out_size; (void)ws_size;
  const float* x      = (const float*)d_in[0];
  const float* rW     = (const float*)d_in[1];
  const float* rb     = (const float*)d_in[2];
  const float* d1W    = (const float*)d_in[3];
  const float* d1b    = (const float*)d_in[4];
  const float* d2W    = (const float*)d_in[5];
  const float* d2b    = (const float*)d_in[6];
  const float* sW_in  = (const float*)d_in[7];
  const float* sb_in  = (const float*)d_in[8];
  const float* sW_cv  = (const float*)d_in[9];
  const float* sb_cv  = (const float*)d_in[10];
  const float* sW_out = (const float*)d_in[11];
  const float* sb_out = (const float*)d_in[12];
  const float* mW     = (const float*)d_in[13];
  const float* mb     = (const float*)d_in[14];
  const float* eW1    = (const float*)d_in[15];
  const float* eb1    = (const float*)d_in[16];
  const float* eW2    = (const float*)d_in[17];
  const float* eb2    = (const float*)d_in[18];
  float* out = (float*)d_out;

  char* p = (char*)d_ws;
  auto alloc = [&](size_t bytes) -> char* {
    char* q = p; p += (bytes + 255) & ~(size_t)255; return q;
  };
  bf16_t* x_bf    = (bf16_t*)alloc((size_t)MTOK * DDIM * 2);
  bf16_t* d1W_bf  = (bf16_t*)alloc((size_t)2 * HDFF * DDIM * 2);
  bf16_t* d2W_bf  = (bf16_t*)alloc((size_t)DDIM * HDFF * 2);
  bf16_t* sWin_bf = (bf16_t*)alloc((size_t)SDIM * DDIM * 2);
  bf16_t* sWcv_bf = (bf16_t*)alloc((size_t)KCONV * SDIM * SDIM * 2);
  bf16_t* sWo_bf  = (bf16_t*)alloc((size_t)DDIM * SDIM * 2);
  bf16_t* eW1_bf  = (bf16_t*)alloc((size_t)NEXP * 2 * HEXP * DDIM * 2);
  bf16_t* eW2_bf  = (bf16_t*)alloc((size_t)NEXP * DDIM * HEXP * 2);
  float*  bw      = (float*) alloc((size_t)MTOK * 3 * 4);
  float*  combine = (float*) alloc((size_t)MTOK * NEXP * 4);
  bf16_t* hdense  = (bf16_t*)alloc((size_t)MTOK * HDFF * 2);
  bf16_t* hin     = (bf16_t*)alloc((size_t)MTOK * SDIM * 2);
  bf16_t* seq     = (bf16_t*)alloc((size_t)MTOK * SDIM * 2);
  bf16_t* hmoe    = (bf16_t*)alloc((size_t)MTOK * HEXP * 2);

  auto cvt = [&](const float* src, bf16_t* dst, long long n) {
    int grid = (int)((n + 255) / 256);
    k_cvt<<<grid, 256, 0, stream>>>(src, dst, n);
  };
  cvt(x,      x_bf,    (long long)MTOK * DDIM);
  cvt(d1W,    d1W_bf,  (long long)2 * HDFF * DDIM);
  cvt(d2W,    d2W_bf,  (long long)DDIM * HDFF);
  cvt(sW_in,  sWin_bf, (long long)SDIM * DDIM);
  cvt(sW_out, sWo_bf,  (long long)DDIM * SDIM);
  cvt(eW1,    eW1_bf,  (long long)NEXP * 2 * HEXP * DDIM);
  cvt(eW2,    eW2_bf,  (long long)NEXP * DDIM * HEXP);
  {
    long long n = (long long)SDIM * SDIM * KCONV;
    k_conv_repack<<<(int)((n + 255) / 256), 256, 0, stream>>>(sW_cv, sWcv_bf, SDIM, KCONV);
  }

  k_router<<<MTOK / 8, 256, 0, stream>>>(x, rW, rb, bw, MTOK, DDIM);
  k_moe_router<<<MTOK / 8, 256, 0, stream>>>(x, mW, mb, bw, combine, MTOK, DDIM);

  dim3 gOut(MTOK / 128, DDIM / 32);   // final projections (N = D), 32x32 per wave

  // branch 0: dense swiglu FFN (accumulate=0 initializes out)
  {
    dim3 g(MTOK / 128, HDFF / 16);
    k_gemm_swiglu<<<g, 128, 0, stream>>>(x_bf, DDIM, d1W_bf, d1b, HDFF, hdense);
    k_gemm_scaled_acc<<<gOut, 128, 0, stream>>>(hdense, HDFF, d2W_bf, d2b, DDIM,
                                                bw, 3, 0, out, 0);
  }
  // branch 1: SSM (TDM-staged in-proj, 4-tap causal conv GEMMs, out-proj)
  {
    dim3 g(MTOK / 128, SDIM / 32);
    k_gemm_bias_tdm<<<g, 128, 0, stream>>>(x_bf, DDIM, sWin_bf, sb_in, SDIM, hin);
    k_gemm_conv<<<g, 128, 0, stream>>>(hin, sWcv_bf, sb_cv, seq, SDIM, TSEQ);
    k_gemm_scaled_acc<<<gOut, 128, 0, stream>>>(seq, SDIM, sWo_bf, sb_out, DDIM,
                                                bw, 3, 1, out, 1);
  }
  // branch 2: MoE dense dispatch (combine already folds bw2; zero off-top-k)
  for (int e = 0; e < NEXP; ++e) {
    dim3 g(MTOK / 128, HEXP / 16);
    k_gemm_swiglu<<<g, 128, 0, stream>>>(x_bf, DDIM,
                                         eW1_bf + (size_t)e * 2 * HEXP * DDIM,
                                         eb1 + (size_t)e * 2 * HEXP, HEXP, hmoe);
    k_gemm_scaled_acc<<<gOut, 128, 0, stream>>>(hmoe, HEXP,
                                                eW2_bf + (size_t)e * DDIM * HEXP,
                                                eb2 + (size_t)e * DDIM, DDIM,
                                                combine, NEXP, e, out, 1);
  }
}
